// FeedforwardEnsemble_61005715472699
// MI455X (gfx1250) — matrix operand order repaired and checked by reading, not compiled, not run on hardware
//
#include <hip/hip_runtime.h>

// ---------------------------------------------------------------------------
// Problem constants (reference: B=4, K=2, S=512, E=8, I=1024, D=2048)
// ---------------------------------------------------------------------------
#define B_ 4
#define K_ 2
#define S_ 512
#define E_ 8
#define I_ 1024
#define D_ 2048
#define NPAIR (B_ * K_ * S_)          // 4096 (token, k) pairs
#define MTILE 64                      // M rows per block / expert padding unit
#define PADROWS (NPAIR + E_ * MTILE)  // 4608

typedef __attribute__((ext_vector_type(16))) __bf16 v16bf;
typedef __attribute__((ext_vector_type(8)))  float  v8f;

__device__ __forceinline__ unsigned short f2bf(float f) {
    unsigned int u = __float_as_uint(f);
    u += 0x7FFFu + ((u >> 16) & 1u);
    return (unsigned short)(u >> 16);
}
// pack two fp32 -> one dword of two bf16 (lo = a, hi = b), RNE
__device__ __forceinline__ unsigned pk2(float a, float b) {
    unsigned ua = __float_as_uint(a);
    ua += 0x7FFFu + ((ua >> 16) & 1u);
    unsigned ub = __float_as_uint(b);
    ub += 0x7FFFu + ((ub >> 16) & 1u);
    return (ua >> 16) | (ub & 0xFFFF0000u);
}
// LDS byte offset of a generic pointer to __shared__ (low 32 bits of flat addr)
__device__ __forceinline__ unsigned lds_off(const void* p) {
    return (unsigned)(size_t)p;
}
// CDNA5 async global->LDS copies (GVS mode: SGPR64 base + per-lane u32 offset)
__device__ __forceinline__ void async_b128(unsigned lds, unsigned goff,
                                           unsigned long long base) {
    asm volatile("global_load_async_to_lds_b128 %0, %1, %2"
                 :: "v"(lds), "v"(goff), "s"(base) : "memory");
}
__device__ __forceinline__ void wait_async0() {
    asm volatile("s_wait_asynccnt 0" ::: "memory");
}

// ---------------------------------------------------------------------------
// Kernel: fp32 -> bf16 bulk conversion, 8 elems/thread, packed b128 stores
// ---------------------------------------------------------------------------
__global__ void __launch_bounds__(256)
cvt_bf16_kernel(const float* __restrict__ src, unsigned short* __restrict__ dst,
                int n8) {
    int i = blockIdx.x * 256 + threadIdx.x;
    if (i >= n8) return;
    const float4* s = (const float4*)src + (size_t)i * 2;
    float4 a = s[0], b = s[1];
    uint4 r = make_uint4(pk2(a.x, a.y), pk2(a.z, a.w),
                         pk2(b.x, b.y), pk2(b.z, b.w));
    ((uint4*)dst)[i] = r;
}

// ---------------------------------------------------------------------------
// Kernel: grouped-GEMM metadata (counts -> 64-padded offsets -> scatter)
// ---------------------------------------------------------------------------
__global__ void __launch_bounds__(256)
prep_kernel(const int* __restrict__ ens,
            int* __restrict__ offsets,        // [E_+1]
            int* __restrict__ pair_token,     // [PADROWS]
            int* __restrict__ pair_pos) {     // [NPAIR]
    __shared__ int cnt[E_];
    __shared__ int cur[E_];
    __shared__ int offs[E_ + 1];
    const int t = threadIdx.x;
    if (t < E_) cnt[t] = 0;
    __syncthreads();
    for (int i = t; i < PADROWS; i += 256) pair_token[i] = 0;
    for (int i = t; i < NPAIR; i += 256) atomicAdd(&cnt[ens[i] & (E_ - 1)], 1);
    __syncthreads();
    if (t == 0) {
        int run = 0;
        for (int e = 0; e < E_; ++e) {
            offs[e] = run;
            cur[e]  = run;
            run += (cnt[e] + (MTILE - 1)) & ~(MTILE - 1);
        }
        offs[E_] = run;
        for (int e = 0; e <= E_; ++e) offsets[e] = offs[e];
    }
    __syncthreads();
    for (int i = t; i < NPAIR; i += 256) {
        int e   = ens[i] & (E_ - 1);
        int pos = atomicAdd(&cur[e], 1);
        int b   = i >> 10;
        int s   = i & (S_ - 1);
        pair_token[pos] = b * S_ + s;
        pair_pos[i]     = pos;
    }
}

// ===========================================================================
// FAST PATH: bf16 sources, async double-buffered staging, explicit 2x
// software pipeline (branchless steady state).
// Block tile: 64 M x 128 N.  Wave: 16-wide N subtile, 4 M subtiles ->
// 4 WMMAs per K-step sharing one B fragment.
// ===========================================================================

// H[p, d] = relu( sum_i W1[e(p), d, i] * x[token(p), i] )     (bf16 out)
__global__ void __launch_bounds__(256)
gemm1_async_kernel(const unsigned short* __restrict__ Xbf,   // (B*S, I)
                   const unsigned short* __restrict__ W1bf,  // (E, D, I)
                   const int* __restrict__ offsets,
                   const int* __restrict__ pair_token,
                   unsigned short* __restrict__ H) {         // (PADROWS, D)
    __shared__ unsigned short At[2][MTILE][32];              // 8 KB
    __shared__ unsigned short Bt[2][128][32];                // 16 KB

    const int rowBase = blockIdx.x * MTILE;
    const int total   = offsets[E_];
    if (rowBase >= total) return;
    int e = 0;
    for (int q = 1; q < E_; ++q) if (rowBase >= offsets[q]) e = q;

    const int n0   = blockIdx.y * 128;
    const int t    = threadIdx.x;
    const int lane = t & 31, wv = t >> 5, half = lane >> 4, lr = lane & 15;

    v8f acc[4] = {{}, {}, {}, {}};

    // staging maps: A = 16 B/thread (1 x b128), B = 32 B/thread (2 x b128)
    const int arow = t >> 2, ak = (t & 3) * 8;
    const int tok  = pair_token[rowBase + arow];
    const unsigned aoff0 = (unsigned)(tok * I_ + ak) * 2u;
    const unsigned long long abase = (unsigned long long)(size_t)Xbf;
    const unsigned alds[2] = { lds_off(&At[0][arow][ak]),
                               lds_off(&At[1][arow][ak]) };

    const int brow = t >> 1, bk = (t & 1) * 16;
    const unsigned boff0 = (unsigned)((n0 + brow) * I_ + bk) * 2u;
    const unsigned long long bbase =
        (unsigned long long)(size_t)(W1bf + (size_t)e * D_ * I_);
    const unsigned blds[2] = { lds_off(&Bt[0][brow][bk]),
                               lds_off(&Bt[1][brow][bk]) };

    auto prefetch = [&](int bf, unsigned kw) {
        async_b128(alds[bf], aoff0 + kw, abase);
        async_b128(blds[bf], boff0 + kw, bbase);
        async_b128(blds[bf] + 16, boff0 + kw + 16, bbase);
    };
    auto step = [&](int bf) {
        union { v16bf v; uint4 q[2]; } a[4], bb;
        bb.q[0] = *(const uint4*)&Bt[bf][wv * 16 + lr][half * 8];
        bb.q[1] = *(const uint4*)&Bt[bf][wv * 16 + lr][16 + half * 8];
#pragma unroll
        for (int m = 0; m < 4; ++m) {
            a[m].q[0] = *(const uint4*)&At[bf][m * 16 + lr][half * 8];
            a[m].q[1] = *(const uint4*)&At[bf][m * 16 + lr][16 + half * 8];
        }
#pragma unroll
        for (int m = 0; m < 4; ++m)
            acc[m] = __builtin_amdgcn_wmma_f32_16x16x32_bf16(
                false, a[m].v, false, bb.v, (short)0, acc[m], false, false);
    };

    const int NIT = I_ / 32;                  // 32 K-steps (even)
    prefetch(0, 0);
    for (int it = 0; it < NIT - 2; it += 2) {
        wait_async0(); __syncthreads();
        prefetch(1, (unsigned)(it + 1) * 64u);
        step(0);
        wait_async0(); __syncthreads();
        prefetch(0, (unsigned)(it + 2) * 64u);
        step(1);
    }
    wait_async0(); __syncthreads();
    prefetch(1, (unsigned)(NIT - 1) * 64u);
    step(0);
    wait_async0(); __syncthreads();
    step(1);

    const int col = n0 + wv * 16 + lr;
#pragma unroll
    for (int m = 0; m < 4; ++m) {
        const int r0 = rowBase + m * 16 + half * 8;
#pragma unroll
        for (int j = 0; j < 8; ++j) {
            float v = acc[m][j] > 0.f ? acc[m][j] : 0.f;
            H[(size_t)(r0 + j) * D_ + col] = f2bf(v);
        }
    }
}

// O[p, i] = sum_d W2[e(p), i, d] * H[p, d]     (fp32 out)
__global__ void __launch_bounds__(256)
gemm2_async_kernel(const unsigned short* __restrict__ H,     // (PADROWS, D)
                   const unsigned short* __restrict__ W2bf,  // (E, I, D)
                   const int* __restrict__ offsets,
                   float* __restrict__ O) {                  // (PADROWS, I)
    __shared__ unsigned short At[2][MTILE][32];
    __shared__ unsigned short Bt[2][128][32];

    const int rowBase = blockIdx.x * MTILE;
    const int total   = offsets[E_];
    if (rowBase >= total) return;
    int e = 0;
    for (int q = 1; q < E_; ++q) if (rowBase >= offsets[q]) e = q;

    const int n0   = blockIdx.y * 128;
    const int t    = threadIdx.x;
    const int lane = t & 31, wv = t >> 5, half = lane >> 4, lr = lane & 15;

    v8f acc[4] = {{}, {}, {}, {}};

    const int arow = t >> 2, ak = (t & 3) * 8;
    const unsigned aoff0 = (unsigned)((rowBase + arow) * D_ + ak) * 2u;
    const unsigned long long abase = (unsigned long long)(size_t)H;
    const unsigned alds[2] = { lds_off(&At[0][arow][ak]),
                               lds_off(&At[1][arow][ak]) };

    const int brow = t >> 1, bk = (t & 1) * 16;
    const unsigned boff0 = (unsigned)((n0 + brow) * D_ + bk) * 2u;
    const unsigned long long bbase =
        (unsigned long long)(size_t)(W2bf + (size_t)e * I_ * D_);
    const unsigned blds[2] = { lds_off(&Bt[0][brow][bk]),
                               lds_off(&Bt[1][brow][bk]) };

    auto prefetch = [&](int bf, unsigned kw) {
        async_b128(alds[bf], aoff0 + kw, abase);
        async_b128(blds[bf], boff0 + kw, bbase);
        async_b128(blds[bf] + 16, boff0 + kw + 16, bbase);
    };
    auto step = [&](int bf) {
        union { v16bf v; uint4 q[2]; } a[4], bb;
        bb.q[0] = *(const uint4*)&Bt[bf][wv * 16 + lr][half * 8];
        bb.q[1] = *(const uint4*)&Bt[bf][wv * 16 + lr][16 + half * 8];
#pragma unroll
        for (int m = 0; m < 4; ++m) {
            a[m].q[0] = *(const uint4*)&At[bf][m * 16 + lr][half * 8];
            a[m].q[1] = *(const uint4*)&At[bf][m * 16 + lr][16 + half * 8];
        }
#pragma unroll
        for (int m = 0; m < 4; ++m)
            acc[m] = __builtin_amdgcn_wmma_f32_16x16x32_bf16(
                false, a[m].v, false, bb.v, (short)0, acc[m], false, false);
    };

    const int NIT = D_ / 32;                  // 64 K-steps (even)
    prefetch(0, 0);
    for (int it = 0; it < NIT - 2; it += 2) {
        wait_async0(); __syncthreads();
        prefetch(1, (unsigned)(it + 1) * 64u);
        step(0);
        wait_async0(); __syncthreads();
        prefetch(0, (unsigned)(it + 2) * 64u);
        step(1);
    }
    wait_async0(); __syncthreads();
    prefetch(1, (unsigned)(NIT - 1) * 64u);
    step(0);
    wait_async0(); __syncthreads();
    step(1);

    const int col = n0 + wv * 16 + lr;
#pragma unroll
    for (int m = 0; m < 4; ++m) {
        const int r0 = rowBase + m * 16 + half * 8;
#pragma unroll
        for (int j = 0; j < 8; ++j)
            O[(size_t)(r0 + j) * I_ + col] = acc[m][j];
    }
}

// ===========================================================================
// FALLBACK PATH (small workspace): convert fp32 -> bf16 while staging,
// packed dword LDS stores to minimize VALU.  Same 64x128 tiling.
// ===========================================================================
__global__ void __launch_bounds__(256)
gemm1_f32_kernel(const float* __restrict__ x,
                 const float* __restrict__ w1,
                 const int* __restrict__ offsets,
                 const int* __restrict__ pair_token,
                 unsigned short* __restrict__ H) {
    __shared__ unsigned short At[MTILE][32];
    __shared__ unsigned short Bt[128][32];

    const int rowBase = blockIdx.x * MTILE;
    const int total   = offsets[E_];
    if (rowBase >= total) return;
    int e = 0;
    for (int q = 1; q < E_; ++q) if (rowBase >= offsets[q]) e = q;

    const int n0   = blockIdx.y * 128;
    const int t    = threadIdx.x;
    const int lane = t & 31, wv = t >> 5, half = lane >> 4, lr = lane & 15;

    v8f acc[4] = {{}, {}, {}, {}};
    const float* w1e = w1 + (size_t)e * D_ * I_;

    const int arow = t >> 2, ak = (t & 3) * 8;
    const int tok  = pair_token[rowBase + arow];
    const float* xrow = x + (size_t)tok * I_;
    const int brow = t >> 1, bk0 = (t & 1) * 16;
    const float* brp = w1e + (size_t)(n0 + brow) * I_;

    for (int kb = 0; kb < I_; kb += 32) {
        __syncthreads();
        {
            const float4* src = (const float4*)(xrow + kb + ak);
            float4 fa = src[0], fb = src[1];
            *(uint4*)&At[arow][ak] =
                make_uint4(pk2(fa.x, fa.y), pk2(fa.z, fa.w),
                           pk2(fb.x, fb.y), pk2(fb.z, fb.w));
        }
        {
            const float4* src = (const float4*)(brp + kb + bk0);
#pragma unroll
            for (int q = 0; q < 2; ++q) {
                float4 fa = src[2 * q], fb = src[2 * q + 1];
                *(uint4*)&Bt[brow][bk0 + q * 8] =
                    make_uint4(pk2(fa.x, fa.y), pk2(fa.z, fa.w),
                               pk2(fb.x, fb.y), pk2(fb.z, fb.w));
            }
        }
        __syncthreads();
        union { v16bf v; uint4 q[2]; } a[4], b;
        b.q[0] = *(const uint4*)&Bt[wv * 16 + lr][half * 8];
        b.q[1] = *(const uint4*)&Bt[wv * 16 + lr][16 + half * 8];
#pragma unroll
        for (int m = 0; m < 4; ++m) {
            a[m].q[0] = *(const uint4*)&At[m * 16 + lr][half * 8];
            a[m].q[1] = *(const uint4*)&At[m * 16 + lr][16 + half * 8];
        }
#pragma unroll
        for (int m = 0; m < 4; ++m)
            acc[m] = __builtin_amdgcn_wmma_f32_16x16x32_bf16(
                false, a[m].v, false, b.v, (short)0, acc[m], false, false);
    }

    const int col = n0 + wv * 16 + lr;
#pragma unroll
    for (int m = 0; m < 4; ++m) {
        const int r0 = rowBase + m * 16 + half * 8;
#pragma unroll
        for (int j = 0; j < 8; ++j) {
            float v = acc[m][j] > 0.f ? acc[m][j] : 0.f;
            H[(size_t)(r0 + j) * D_ + col] = f2bf(v);
        }
    }
}

__global__ void __launch_bounds__(256)
gemm2_f32_kernel(const unsigned short* __restrict__ H,
                 const float* __restrict__ w2,
                 const int* __restrict__ offsets,
                 float* __restrict__ O) {
    __shared__ unsigned short At[MTILE][32];
    __shared__ unsigned short Bt[128][32];

    const int rowBase = blockIdx.x * MTILE;
    const int total   = offsets[E_];
    if (rowBase >= total) return;
    int e = 0;
    for (int q = 1; q < E_; ++q) if (rowBase >= offsets[q]) e = q;

    const int n0   = blockIdx.y * 128;
    const int t    = threadIdx.x;
    const int lane = t & 31, wv = t >> 5, half = lane >> 4, lr = lane & 15;

    v8f acc[4] = {{}, {}, {}, {}};
    const float* w2e = w2 + (size_t)e * I_ * D_;

    const int arow = t >> 2, ak = (t & 3) * 8;
    const unsigned short* hrow = H + (size_t)(rowBase + arow) * D_;
    const int brow = t >> 1, bk0 = (t & 1) * 16;
    const float* brp = w2e + (size_t)(n0 + brow) * D_;

    for (int kb = 0; kb < D_; kb += 32) {
        __syncthreads();
        *(uint4*)&At[arow][ak] = *(const uint4*)(hrow + kb + ak);
        {
            const float4* src = (const float4*)(brp + kb + bk0);
#pragma unroll
            for (int q = 0; q < 2; ++q) {
                float4 fa = src[2 * q], fb = src[2 * q + 1];
                *(uint4*)&Bt[brow][bk0 + q * 8] =
                    make_uint4(pk2(fa.x, fa.y), pk2(fa.z, fa.w),
                               pk2(fb.x, fb.y), pk2(fb.z, fb.w));
            }
        }
        __syncthreads();
        union { v16bf v; uint4 q[2]; } a[4], b;
        b.q[0] = *(const uint4*)&Bt[wv * 16 + lr][half * 8];
        b.q[1] = *(const uint4*)&Bt[wv * 16 + lr][16 + half * 8];
#pragma unroll
        for (int m = 0; m < 4; ++m) {
            a[m].q[0] = *(const uint4*)&At[m * 16 + lr][half * 8];
            a[m].q[1] = *(const uint4*)&At[m * 16 + lr][16 + half * 8];
        }
#pragma unroll
        for (int m = 0; m < 4; ++m)
            acc[m] = __builtin_amdgcn_wmma_f32_16x16x32_bf16(
                false, a[m].v, false, b.v, (short)0, acc[m], false, false);
    }

    const int col = n0 + wv * 16 + lr;
#pragma unroll
    for (int m = 0; m < 4; ++m) {
        const int r0 = rowBase + m * 16 + half * 8;
#pragma unroll
        for (int j = 0; j < 8; ++j)
            O[(size_t)(r0 + j) * I_ + col] = acc[m][j];
    }
}

// ---------------------------------------------------------------------------
// Kernel: out[b,s,i] = w0 * O[p0, i] + w1 * O[p1, i]
// ---------------------------------------------------------------------------
__global__ void __launch_bounds__(256)
combine_kernel(const float* __restrict__ weights,
               const int*   __restrict__ pair_pos,
               const float* __restrict__ O,
               float* __restrict__ out) {
    const int tkn = blockIdx.x;
    const int b   = tkn >> 9;
    const int s   = tkn & (S_ - 1);
    const int i0  = threadIdx.x * 4;
    const int idx0 = (b * K_ + 0) * S_ + s;
    const int idx1 = (b * K_ + 1) * S_ + s;
    const float w0 = weights[idx0], w1 = weights[idx1];
    const int   p0 = pair_pos[idx0], p1 = pair_pos[idx1];
    float4 o0 = *(const float4*)(O + (size_t)p0 * I_ + i0);
    float4 o1 = *(const float4*)(O + (size_t)p1 * I_ + i0);
    float4 r;
    r.x = w0 * o0.x + w1 * o1.x;
    r.y = w0 * o0.y + w1 * o1.y;
    r.z = w0 * o0.z + w1 * o1.z;
    r.w = w0 * o0.w + w1 * o1.w;
    *(float4*)(out + (size_t)tkn * I_ + i0) = r;
}

// ---------------------------------------------------------------------------
// Host launcher
// ---------------------------------------------------------------------------
extern "C" void kernel_launch(void* const* d_in, const int* in_sizes, int n_in,
                              void* d_out, int out_size, void* d_ws, size_t ws_size,
                              hipStream_t stream) {
    (void)in_sizes; (void)n_in; (void)out_size;

    const float* x        = (const float*)d_in[0];   // (B,S,I)
    const float* weights  = (const float*)d_in[1];   // (B,K,S)
    const int*   ens      = (const int*)  d_in[2];   // (B,K,S)
    const float* proj_in  = (const float*)d_in[3];   // (E,D,I)
    const float* proj_out = (const float*)d_in[4];   // (E,I,D)
    float*       out      = (float*)d_out;           // (B,S,I)

    // workspace layout
    char* ws = (char*)d_ws;
    size_t off_offsets = 0;
    size_t off_ptok    = 256;
    size_t off_ppos    = 256 + ((PADROWS * 4 + 255) & ~255ull);
    size_t off_H       = (off_ppos + NPAIR * 4 + 255) & ~255ull;
    size_t off_O       = off_H  + (size_t)PADROWS * D_ * 2;          // H: 18.9 MB
    size_t off_X       = off_O  + (size_t)PADROWS * I_ * 4;          // O: 18.9 MB
    size_t off_W1      = off_X  + (size_t)B_ * S_ * I_ * 2;          // X: 4.2 MB
    size_t off_W2      = off_W1 + (size_t)E_ * D_ * I_ * 2;          // W1: 33.5 MB
    size_t need_full   = off_W2 + (size_t)E_ * I_ * D_ * 2;          // ~109 MB

    int*            offsets    = (int*)(ws + off_offsets);
    int*            pair_token = (int*)(ws + off_ptok);
    int*            pair_pos   = (int*)(ws + off_ppos);
    unsigned short* H          = (unsigned short*)(ws + off_H);
    float*          O          = (float*)(ws + off_O);
    unsigned short* Xbf        = (unsigned short*)(ws + off_X);
    unsigned short* W1bf       = (unsigned short*)(ws + off_W1);
    unsigned short* W2bf       = (unsigned short*)(ws + off_W2);

    prep_kernel<<<1, 256, 0, stream>>>(ens, offsets, pair_token, pair_pos);

    dim3 g1(PADROWS / MTILE, D_ / 128);   // (72, 16)
    dim3 g2(PADROWS / MTILE, I_ / 128);   // (72, 8)

    if (ws_size >= need_full) {
        int nw = (E_ * D_ * I_) / 8;
        int nx = (B_ * S_ * I_) / 8;
        cvt_bf16_kernel<<<(nw + 255) / 256, 256, 0, stream>>>(proj_in,  W1bf, nw);
        cvt_bf16_kernel<<<(nw + 255) / 256, 256, 0, stream>>>(proj_out, W2bf, nw);
        cvt_bf16_kernel<<<(nx + 255) / 256, 256, 0, stream>>>(x,        Xbf,  nx);

        gemm1_async_kernel<<<g1, 256, 0, stream>>>(Xbf, W1bf, offsets, pair_token, H);
        gemm2_async_kernel<<<g2, 256, 0, stream>>>(H, W2bf, offsets, O);
    } else {
        gemm1_f32_kernel<<<g1, 256, 0, stream>>>(x, proj_in, offsets, pair_token, H);
        gemm2_f32_kernel<<<g2, 256, 0, stream>>>(H, proj_out, offsets, O);
    }

    combine_kernel<<<B_ * S_, 256, 0, stream>>>(weights, pair_pos, O, out);
}
